// MyModel_87522843559427
// MI455X (gfx1250) — matrix operand order repaired and checked by reading, not compiled, not run on hardware
//
#include <hip/hip_runtime.h>
#include <hip/hip_bf16.h>

// ---------------------------------------------------------------------------
// Reference: conv3x3(3->16)+ReLU, conv1x1(16->17), greedy NMS (5 picks) over
// batch 0 of a 4x1024x1024x3 int image. Output (1,5,17) f32 = 85 floats.
// Only batch 0 affects the output -> compute batch 0 only.
// Conv1/Conv2 done as V_WMMA_F32_16X16X32_F16 (K=27 / K=16 zero-padded to 32).
// All WMMA operands are built from LDS (weights staged zero-padded + transposed)
// so no exec-masked global loads appear in the inner path.
// ---------------------------------------------------------------------------

typedef __attribute__((ext_vector_type(16))) _Float16 v16h;
typedef __attribute__((ext_vector_type(8)))  _Float16 v8h;
typedef __attribute__((ext_vector_type(8)))  float    v8f;

#define IMG_W 1024
#define IMG_H 1024
#define NPIX  (IMG_W * IMG_H)
#define NEG_INF_F (-1e30f)
#define SCORE_THR 0.75f
#define IOU_THR   0.5f
#define MAX_OUT   5

// ---- monotone u64 key: higher score wins, ties -> lower index wins ---------
__device__ __forceinline__ unsigned int f32_orderable(float f) {
    unsigned int u = __float_as_uint(f);
    return (u & 0x80000000u) ? ~u : (u | 0x80000000u);
}
__device__ __forceinline__ float f32_unorder(unsigned int u) {
    return __uint_as_float((u & 0x80000000u) ? (u ^ 0x80000000u) : ~u);
}
__device__ __forceinline__ unsigned long long make_key(float s, unsigned int idx) {
    return ((unsigned long long)f32_orderable(s) << 32) |
           (unsigned long long)(0xFFFFFFFFu - idx);
}

// ---------------------------------------------------------------------------
// Block = 256 thr = 8 waves; each wave owns 16 consecutive pixels of one image
// row; block covers 128 pixels.  Two WMMAs per wave.
// ---------------------------------------------------------------------------
__global__ __launch_bounds__(256)
void conv_box_kernel(const int* __restrict__ in,
                     const float* __restrict__ w1, const float* __restrict__ b1,
                     const float* __restrict__ w2, const float* __restrict__ b2,
                     float* __restrict__ scores, float4* __restrict__ boxes)
{
    __shared__ float tile[3][130][3];                 // rows y-1..y+1, halo cols
    __shared__ __align__(32) _Float16 w1t[16][32];    // [n][K], K>=27 zero
    __shared__ __align__(32) _Float16 w2t[16][32];    // [n][K], K>=16 zero
    __shared__ __align__(32) _Float16 t_lds[8][16][32]; // [wave][pixel][K] f16, K>=16 zero
    __shared__ float w2s[16];                         // W2 score column
    __shared__ float b1s[16], b2s[16];
    __shared__ float b2_score;
    __shared__ float o_lds[8][16][4];                 // conv2 out ch0..3
    __shared__ float s_lds[8][16];                    // score channel

    const int tid = threadIdx.x;
    const int y   = blockIdx.x >> 3;                  // 1024 rows
    const int x0  = (blockIdx.x & 7) << 7;            // 8 blocks/row * 128 px

    // ---- stage weights (zero-padded, transposed) + biases ------------------
    for (int i = tid; i < 16 * 32; i += 256) {
        int n = i >> 5, K = i & 31;
        w1t[n][K] = (_Float16)((K < 27) ? w1[K * 16 + n] : 0.0f);
        w2t[n][K] = (_Float16)((K < 16) ? w2[K * 17 + n] : 0.0f);
    }
    if (tid < 16) {
        w2s[tid] = w2[tid * 17 + 16];
        b1s[tid] = b1[tid];
        b2s[tid] = b2[tid];
    }
    if (tid == 0) b2_score = b2[16];
    // zero the K=16..31 pad of t_lds (one row of 16 halves per thread)
    if (tid < 128) {
#pragma unroll
        for (int k = 16; k < 32; ++k) t_lds[tid >> 4][tid & 15][k] = (_Float16)0.0f;
    }

    // ---- load input tile with zero halo (normalize int -> [-1,1]) ----------
    for (int i = tid; i < 3 * 130 * 3; i += 256) {
        int c   = i % 3;
        int col = (i / 3) % 130;
        int row = i / (3 * 130);
        int gy = y + row - 1;
        int gx = x0 + col - 1;
        float v = 0.0f;
        if (gy >= 0 && gy < IMG_H && gx >= 0 && gx < IMG_W)
            v = (float)in[(gy * IMG_W + gx) * 3 + c] * (1.0f / 127.5f) - 1.0f;
        tile[row][col][c] = v;
    }
    __syncthreads();

    const int wave = tid >> 5;
    const int lane = tid & 31;
    const int g    = lane >> 4;        // half-wave group
    const int m    = lane & 15;        // M row (pixel) / N col index
    const int pxl  = (wave << 4) + m;  // pixel within block

    // ---- conv1: A = im2col patch (16x32, K=27 padded) ----------------------
    v16h a1;
#pragma unroll
    for (int hi = 0; hi < 16; ++hi) {
        int K = hi + 16 * g;                               // 16-bit A layout (contiguous)
        int Kc = (K < 27) ? K : 0;
        int ky = Kc / 9, r = Kc % 9, kx = r / 3, ci = r % 3;
        float v = tile[ky][pxl + kx][ci];                  // LDS, no fault risk
        a1[hi] = (_Float16)((K < 27) ? v : 0.0f);
    }
    // B operand: contiguous 16 halves from the transposed LDS panel
    v16h b1m = *(const v16h*)&w1t[m][16 * g];

    v8f acc1;
    {
        float bias = b1s[m];                               // C[m][n] = b1[n]
#pragma unroll
        for (int r = 0; r < 8; ++r) acc1[r] = bias;
    }
    acc1 = __builtin_amdgcn_wmma_f32_16x16x32_f16(false, a1, false, b1m,
                                                  (short)0, acc1, false, false);

    // ReLU + restripe D (lane=channel n, row r=pixel r+8g) into [pixel][K] f16
#pragma unroll
    for (int r = 0; r < 8; ++r)
        t_lds[wave][r + 8 * g][m] = (_Float16)fmaxf(acc1[r], 0.0f);
    __syncthreads();

    // ---- conv2: A = t (16x16 pad to K=32), B = W2[:, 0..15] ----------------
    // A halves 0..7  <- K = 16g + 0..7 ; halves 8..15 <- K = 16g + 8 + 0..7
    v8h a2lo = *(const v8h*)&t_lds[wave][m][16 * g];
    v8h a2hi = *(const v8h*)&t_lds[wave][m][16 * g + 8];
    v16h a2 = __builtin_shufflevector(a2lo, a2hi,
                                      0, 1, 2, 3, 4, 5, 6, 7,
                                      8, 9, 10, 11, 12, 13, 14, 15);
    v16h b2m = *(const v16h*)&w2t[m][16 * g];

    v8f acc2;
    {
        float bias = b2s[m];
#pragma unroll
        for (int r = 0; r < 8; ++r) acc2[r] = bias;
    }
    acc2 = __builtin_amdgcn_wmma_f32_16x16x32_f16(false, a2, false, b2m,
                                                  (short)0, acc2, false, false);

    if (m < 4) {
#pragma unroll
        for (int r = 0; r < 8; ++r) o_lds[wave][r + 8 * g][m] = acc2[r];
    }
    // score channel 16 via VALU dot (lanes 0..15 each own one pixel, LDS only)
    if (g == 0) {
        float s = b2_score;
#pragma unroll
        for (int k = 0; k < 16; ++k)
            s += (float)t_lds[wave][m][k] * w2s[k];
        s_lds[wave][m] = s;
    }
    __syncthreads();

    // ---- per-pixel box + thresholded score --------------------------------
    if (tid < 128) {
        int wv = tid >> 4, pm = tid & 15;
        float cy = o_lds[wv][pm][0];
        float cx = o_lds[wv][pm][1];
        float hh = o_lds[wv][pm][2];
        float ww = o_lds[wv][pm][3];
        float s  = s_lds[wv][pm];
        float tx = fminf(fmaxf(cx - 0.5f * ww, 0.0f), 1e8f);
        float ty = fminf(fmaxf(cy - 0.5f * hh, 0.0f), 1e8f);
        float bx = cx + 0.5f * ww;
        float by = cy + 0.5f * hh;
        int gid = y * IMG_W + x0 + tid;
        boxes[gid]  = make_float4(tx, ty, bx, by);
        scores[gid] = (s > SCORE_THR) ? s : NEG_INF_F;
    }
}

// ---------------------------------------------------------------------------
__global__ void init_kernel(unsigned long long* best) {
    if (threadIdx.x < MAX_OUT) best[threadIdx.x] = 0ull;
}

__global__ __launch_bounds__(256)
void argmax_kernel(const float* __restrict__ scores,
                   unsigned long long* __restrict__ best, int iter)
{
    __shared__ unsigned long long red[256];
    unsigned long long lk = 0ull;
    for (unsigned int gid = blockIdx.x * 256u + threadIdx.x; gid < NPIX;
         gid += gridDim.x * 256u) {
        unsigned long long k = make_key(scores[gid], gid);
        lk = (k > lk) ? k : lk;
    }
    red[threadIdx.x] = lk;
    __syncthreads();
    for (int off = 128; off > 0; off >>= 1) {
        if (threadIdx.x < off) {
            unsigned long long o = red[threadIdx.x + off];
            if (o > red[threadIdx.x]) red[threadIdx.x] = o;
        }
        __syncthreads();
    }
    if (threadIdx.x == 0) atomicMax(&best[iter], red[0]);
}

__global__ __launch_bounds__(256)
void suppress_kernel(float* __restrict__ scores,
                     const float4* __restrict__ boxes,
                     const unsigned long long* __restrict__ best, int iter)
{
    unsigned long long k = best[iter];
    unsigned int idx = 0xFFFFFFFFu - (unsigned int)(k & 0xFFFFFFFFull);
    float4 bb = boxes[idx];
    float area_a = (bb.z - bb.x) * (bb.w - bb.y);
    for (unsigned int gid = blockIdx.x * 256u + threadIdx.x; gid < NPIX;
         gid += gridDim.x * 256u) {
        float4 o = boxes[gid];
        float x1 = fmaxf(bb.x, o.x), y1 = fmaxf(bb.y, o.y);
        float x2 = fminf(bb.z, o.z), y2 = fminf(bb.w, o.w);
        float inter = fmaxf(x2 - x1, 0.0f) * fmaxf(y2 - y1, 0.0f);
        float area_b = (o.z - o.x) * (o.w - o.y);
        float un = area_a + area_b - inter;
        float iou = (un > 0.0f) ? inter / un : 0.0f;
        if (iou > IOU_THR || gid == idx) scores[gid] = NEG_INF_F;
    }
}

// ---------------------------------------------------------------------------
// Recompute the 5 winning rows in f32 from raw input (cheap; avoids storing
// the full 1M x 17 feature map). out[j*17+c]; ch 0..15 scaled by 128.
// ---------------------------------------------------------------------------
__global__ void gather_kernel(const int* __restrict__ in,
                              const float* __restrict__ w1, const float* __restrict__ b1,
                              const float* __restrict__ w2, const float* __restrict__ b2,
                              const unsigned long long* __restrict__ best,
                              float* __restrict__ out)
{
    int t = threadIdx.x;
    if (t >= MAX_OUT * 17) return;
    int j = t / 17, c = t % 17;
    unsigned long long k = best[j];
    unsigned int idx = 0xFFFFFFFFu - (unsigned int)(k & 0xFFFFFFFFull);
    float s_sel = f32_unorder((unsigned int)(k >> 32));
    bool valid = s_sel > (NEG_INF_F * 0.5f);
    float result = 0.0f;
    if (valid) {
        int y = idx >> 10, x = idx & 1023;
        float t16[16];
#pragma unroll
        for (int kk = 0; kk < 16; ++kk) t16[kk] = b1[kk];
        for (int K = 0; K < 27; ++K) {
            int ky = K / 9, r = K % 9, kx = r / 3, ci = r % 3;
            int gy = y + ky - 1, gx = x + kx - 1;
            float pv = 0.0f;
            if (gy >= 0 && gy < IMG_H && gx >= 0 && gx < IMG_W)
                pv = (float)in[(gy * IMG_W + gx) * 3 + ci] * (1.0f / 127.5f) - 1.0f;
#pragma unroll
            for (int kk = 0; kk < 16; ++kk) t16[kk] += pv * w1[K * 16 + kk];
        }
        float val = b2[c];
#pragma unroll
        for (int kk = 0; kk < 16; ++kk) val += fmaxf(t16[kk], 0.0f) * w2[kk * 17 + c];
        result = (c == 16) ? val : val * 128.0f;
    }
    out[t] = result;
}

// ---------------------------------------------------------------------------
extern "C" void kernel_launch(void* const* d_in, const int* in_sizes, int n_in,
                              void* d_out, int out_size, void* d_ws, size_t ws_size,
                              hipStream_t stream)
{
    (void)in_sizes; (void)n_in; (void)out_size; (void)ws_size;
    const int*   in = (const int*)  d_in[0];   // 4x1024x1024x3 (batch 0 used)
    const float* w1 = (const float*)d_in[1];   // 27x16
    const float* b1 = (const float*)d_in[2];   // 16
    const float* w2 = (const float*)d_in[3];   // 16x17
    const float* b2 = (const float*)d_in[4];   // 17
    float* out = (float*)d_out;                // 85 floats

    char* ws = (char*)d_ws;                    // needs ~20 MB
    float*  scores = (float*) ws;                                   //  4 MB
    float4* boxes  = (float4*)(ws + (size_t)NPIX * sizeof(float));  // 16 MB
    unsigned long long* best =
        (unsigned long long*)(ws + (size_t)NPIX * sizeof(float)
                                 + (size_t)NPIX * sizeof(float4)); // 40 B

    init_kernel<<<1, 32, 0, stream>>>(best);
    conv_box_kernel<<<(IMG_H * IMG_W) / 128, 256, 0, stream>>>(
        in, w1, b1, w2, b2, scores, boxes);
    for (int i = 0; i < MAX_OUT; ++i) {
        argmax_kernel<<<512, 256, 0, stream>>>(scores, best, i);
        if (i < MAX_OUT - 1)
            suppress_kernel<<<512, 256, 0, stream>>>(scores, boxes, best, i);
    }
    gather_kernel<<<1, 128, 0, stream>>>(in, w1, b1, w2, b2, best, out);
}
